// LatentDiffiT_1314259993194
// MI455X (gfx1250) — matrix operand order repaired and checked by reading, not compiled, not run on hardware
//
#include <hip/hip_runtime.h>
#include <hip/hip_bf16.h>
#include <math.h>
#include <stdint.h>

// ---------------------------------------------------------------------------
// DiT forward (LatentDiffiT): B=32, N=256, D=768, L=12, H=12, HD=64, MLP=3072
// Heavy GEMMs: v_wmma_f32_16x16x32_bf16 (bf16 in, f32 accum).
// Weight-GEMMs double-buffer LDS; A tile staged by the Tensor Data Mover
// (tensor_load_to_lds + s_wait_tensorcnt) when the builtin is available.
// ---------------------------------------------------------------------------

typedef __attribute__((ext_vector_type(16))) __bf16 v16bf;
typedef __attribute__((ext_vector_type(8)))  float  v8f;
typedef __attribute__((ext_vector_type(4)))  unsigned int u32x4;
typedef __attribute__((ext_vector_type(8)))  int          i32x8;
typedef __attribute__((ext_vector_type(4)))  int          i32x4;

#if __has_builtin(__builtin_amdgcn_tensor_load_to_lds)
#define HAVE_TDM 1
#else
#define HAVE_TDM 0
#endif

struct U8 { unsigned int u[8]; };

static __device__ __forceinline__ unsigned short f2bf(float f) {
  unsigned int u = __builtin_bit_cast(unsigned int, f);
  unsigned int r = u + 0x7FFFu + ((u >> 16) & 1u);   // round-to-nearest-even
  return (unsigned short)(r >> 16);
}
static __device__ __forceinline__ float bf2f(unsigned short h) {
  unsigned int u = ((unsigned int)h) << 16;
  return __builtin_bit_cast(float, u);
}
static __device__ __forceinline__ unsigned int pack2(unsigned short a, unsigned short b) {
  return (unsigned int)a | ((unsigned int)b << 16);
}
// Low 32 bits of a generic pointer to LDS == workgroup-relative LDS byte addr
static __device__ __forceinline__ unsigned int lds_addr32(const void* p) {
  return (unsigned int)(uintptr_t)p;
}

// ---------------------------------------------------------------------------
// Weight GEMM: C[M,N] = A[M,K](bf16) * W[K,N](f32->bf16), tile 128x64, KC=32.
// 8 waves, each computing a 32x32 sub-block (2x2 WMMA 16x16 tiles).
// Double-buffered LDS; A tile via TDM (wave 0 issues, EXEC-independent).
// MODE 0: out bf16 = acc + bias[n] + rowbias[batch][n]
// MODE 1: out f32 += acc + bias[n]                     (residual accumulate)
// MODE 2: out bf16 = silu(acc + bias[n])
// ---------------------------------------------------------------------------
template<int MODE>
__global__ void __launch_bounds__(256) gemm_wmma_w(
    const unsigned short* __restrict__ A, const float* __restrict__ Bw,
    const float* __restrict__ bias, const float* __restrict__ rowbias,
    void* __restrict__ Out, int K, int lda, int ldb, int ldo)
{
  __shared__ unsigned int sA[2][128][16];   // 2 x (128 rows x 16 bf16-pairs)
  __shared__ unsigned int sB[2][16][64];    // 2 x (16 pair-rows x 64 cols)

  const int tid    = threadIdx.x;
  const int wid    = tid >> 5;
  const int lane   = tid & 31;
  const int lane16 = lane & 15;
  const int hi     = lane >> 4;
  const int warpM  = (wid & 3) * 32;
  const int warpN  = (wid >> 2) * 32;
  const int mbase  = blockIdx.x * 128;
  const int nbase  = blockIdx.y * 64;
  const int nchunk = K >> 5;

#if HAVE_TDM
  // Issue a TDM 2D tile load: 128 rows x 64B (16 u32) into sA[slot].
  auto tdm_issue = [&](int c, int slot) {
    unsigned long long ga = (unsigned long long)(uintptr_t)A
                          + ((unsigned long long)mbase * (unsigned long long)lda
                             + (unsigned long long)c * 32ull) * 2ull;
    unsigned int lp = lds_addr32(&sA[slot][0][0]);
    unsigned int stride0 = (unsigned int)(lda >> 1);     // row stride in u32
    u32x4 g0; i32x8 g1;
    i32x4 g2 = {0, 0, 0, 0};
    i32x4 g3 = {0, 0, 0, 0};
    g0[0] = 1u;                                          // count=1, user mode
    g0[1] = lp;                                          // lds_addr
    g0[2] = (unsigned int)(ga & 0xffffffffull);          // global_addr[31:0]
    g0[3] = (unsigned int)((ga >> 32) & 0x1ffffffull) | (2u << 30); // [56:32]|type=2
    g1[0] = (int)(2u << 16);        // wg_mask=0, data_size=2 (4B elements)
    g1[1] = (int)(16u << 16);       // tensor_dim0[15:0]=16 (bits 63:48)
    g1[2] = (int)(128u << 16);      // tensor_dim1[15:0]=128 (bits 95:80)
    g1[3] = (int)(16u << 16);       // tile_dim0=16 (bits 127:112)
    g1[4] = (int)128u;              // tile_dim1=128 (bits 143:128)
    g1[5] = (int)stride0;           // tensor_dim0_stride[31:0]
    g1[6] = 0;
    g1[7] = 0;
#if defined(__clang_major__) && (__clang_major__ >= 23)
    i32x8 g4 = {0, 0, 0, 0, 0, 0, 0, 0};
    __builtin_amdgcn_tensor_load_to_lds(g0, g1, g2, g3, g4, 0);
#else
    __builtin_amdgcn_tensor_load_to_lds(g0, g1, g2, g3, 0);
#endif
  };
#else
  unsigned int areg[8];
  auto aload = [&](int c) {
    #pragma unroll
    for (int s = 0; s < 8; ++s) {
      int i = tid + s * 256; int r = i >> 4, cc = i & 15;
      areg[s] = *(const unsigned int*)(A + (size_t)(mbase + r) * lda + c * 32 + 2 * cc);
    }
  };
  auto astore = [&](int slot) {
    #pragma unroll
    for (int s = 0; s < 8; ++s) {
      int i = tid + s * 256; int r = i >> 4, cc = i & 15;
      sA[slot][r][cc] = areg[s];
    }
  };
#endif

  float breg[4][2];
  auto bload = [&](int c) {
    #pragma unroll
    for (int s = 0; s < 4; ++s) {
      int i = tid + s * 256; int kp = i >> 6, n = i & 63;
      int gk = c * 32 + 2 * kp, gn = nbase + n;
      breg[s][0] = Bw[(size_t)gk * ldb + gn];
      breg[s][1] = Bw[(size_t)(gk + 1) * ldb + gn];
    }
  };
  auto bstore = [&](int slot) {
    #pragma unroll
    for (int s = 0; s < 4; ++s) {
      int i = tid + s * 256; int kp = i >> 6, n = i & 63;
      sB[slot][kp][n] = pack2(f2bf(breg[s][0]), f2bf(breg[s][1]));
    }
  };

  v8f zero = {0.f,0.f,0.f,0.f,0.f,0.f,0.f,0.f};
  v8f acc[2][2];
  acc[0][0] = zero; acc[0][1] = zero; acc[1][0] = zero; acc[1][1] = zero;

  // ---- prologue: stage chunk 0 into slot 0
#if HAVE_TDM
  if (wid == 0) tdm_issue(0, 0);
#else
  aload(0);
#endif
  bload(0);
#if !HAVE_TDM
  astore(0);
#endif
  bstore(0);
#if HAVE_TDM
  if (wid == 0) __builtin_amdgcn_s_wait_tensorcnt(0);
#endif
  __syncthreads();

  for (int c = 0; c < nchunk; ++c) {
    const int buf = c & 1, nxt = buf ^ 1;
    const bool more = (c + 1) < nchunk;
    if (more) {
#if HAVE_TDM
      if (wid == 0) tdm_issue(c + 1, nxt);   // DMA overlaps this chunk's WMMAs
#else
      aload(c + 1);
#endif
      bload(c + 1);
    }

    // ---- compute on slot `buf`
    U8 au[2], bu[2];
    #pragma unroll
    for (int mt = 0; mt < 2; ++mt) {
      int r = warpM + mt * 16 + lane16;
      #pragma unroll
      for (int j = 0; j < 4; ++j) {
        au[mt].u[j]     = sA[buf][r][hi * 4 + j];
        au[mt].u[4 + j] = sA[buf][r][8 + hi * 4 + j];
      }
    }
    #pragma unroll
    for (int nt = 0; nt < 2; ++nt) {
      int cc = warpN + nt * 16 + lane16;
      #pragma unroll
      for (int j = 0; j < 8; ++j) bu[nt].u[j] = sB[buf][(j < 4 ? hi * 4 + j : 4 + hi * 4 + j)][cc];
    }
    #pragma unroll
    for (int mt = 0; mt < 2; ++mt) {
      v16bf av = __builtin_bit_cast(v16bf, au[mt]);
      #pragma unroll
      for (int nt = 0; nt < 2; ++nt) {
        v16bf bv = __builtin_bit_cast(v16bf, bu[nt]);
        acc[mt][nt] = __builtin_amdgcn_wmma_f32_16x16x32_bf16(
            false, av, false, bv, (short)0, acc[mt][nt], false, false);
      }
    }

    if (more) {
#if !HAVE_TDM
      astore(nxt);
#endif
      bstore(nxt);
#if HAVE_TDM
      if (wid == 0) __builtin_amdgcn_s_wait_tensorcnt(0);
#endif
      __syncthreads();
    }
  }

  // ---- epilogue (C layout: VGPR r -> M = r + hi*8; lane16 -> N)
  #pragma unroll
  for (int mt = 0; mt < 2; ++mt)
    #pragma unroll
    for (int nt = 0; nt < 2; ++nt)
      #pragma unroll
      for (int r = 0; r < 8; ++r) {
        int gm = mbase + warpM + mt * 16 + hi * 8 + r;
        int gn = nbase + warpN + nt * 16 + lane16;
        float v = acc[mt][nt][r];
        if constexpr (MODE == 0) {
          v += bias[gn] + rowbias[(size_t)(gm >> 8) * 768 + gn];
          ((unsigned short*)Out)[(size_t)gm * ldo + gn] = f2bf(v);
        } else if constexpr (MODE == 1) {
          ((float*)Out)[(size_t)gm * ldo + gn] += v + bias[gn];
        } else {
          v += bias[gn];
          float s = v / (1.0f + __expf(-v));
          ((unsigned short*)Out)[(size_t)gm * ldo + gn] = f2bf(s);
        }
      }
}

// ---------------------------------------------------------------------------
// Batched attention GEMMs (per b,h via blockIdx.z), single-buffered.
// MODE 3: scores = Q * K^T           -> f32 [bh,256,256]
// MODE 4: out    = P * V             -> bf16 strided into [B,N,D]
// ---------------------------------------------------------------------------
template<int MODE>
__global__ void __launch_bounds__(256) gemm_wmma_att(
    const unsigned short* __restrict__ A, const unsigned short* __restrict__ Bm,
    void* __restrict__ Out, int K, int lda, int ldb, int ldo)
{
  __shared__ unsigned int sA[128][16];
  __shared__ unsigned int sB[16][64];

  const int tid    = threadIdx.x;
  const int wid    = tid >> 5;
  const int lane   = tid & 31;
  const int lane16 = lane & 15;
  const int hi     = lane >> 4;
  const int warpM  = (wid & 3) * 32;
  const int warpN  = (wid >> 2) * 32;
  const int mbase  = blockIdx.x * 128;
  const int nbase  = blockIdx.y * 64;

  const int bh = blockIdx.z, b = bh / 12, h = bh % 12;
  size_t aoff, boff, ooff;
  if constexpr (MODE == 3) {
    aoff = (size_t)b * (256 * 768) + (size_t)h * 64;
    boff = aoff;
    ooff = (size_t)bh * (256 * 256);
  } else {
    aoff = (size_t)bh * (256 * 256);
    boff = (size_t)b * (256 * 768) + (size_t)h * 64;
    ooff = boff;
  }

  v8f zero = {0.f,0.f,0.f,0.f,0.f,0.f,0.f,0.f};
  v8f acc[2][2];
  acc[0][0] = zero; acc[0][1] = zero; acc[1][0] = zero; acc[1][1] = zero;

  for (int kk = 0; kk < K; kk += 32) {
    for (int i = tid; i < 128 * 16; i += 256) {
      int r = i >> 4, c = i & 15;
      sA[r][c] = *(const unsigned int*)(A + aoff + (size_t)(mbase + r) * lda + kk + 2 * c);
    }
    if constexpr (MODE == 3) {
      // B[k][n] = Kmat[n][k]: pairs over k contiguous within Kmat row n
      for (int i = tid; i < 16 * 64; i += 256) {
        int kp = i >> 6, n = i & 63;
        sB[kp][n] = *(const unsigned int*)(Bm + boff + (size_t)(nbase + n) * ldb + kk + 2 * kp);
      }
    } else {
      for (int i = tid; i < 16 * 64; i += 256) {
        int kp = i >> 6, n = i & 63;
        int gk = kk + 2 * kp, gn = nbase + n;
        unsigned short h0 = Bm[boff + (size_t)gk * ldb + gn];
        unsigned short h1 = Bm[boff + (size_t)(gk + 1) * ldb + gn];
        sB[kp][n] = pack2(h0, h1);
      }
    }
    __syncthreads();

    U8 au[2], bu[2];
    #pragma unroll
    for (int mt = 0; mt < 2; ++mt) {
      int r = warpM + mt * 16 + lane16;
      #pragma unroll
      for (int j = 0; j < 4; ++j) {
        au[mt].u[j]     = sA[r][hi * 4 + j];
        au[mt].u[4 + j] = sA[r][8 + hi * 4 + j];
      }
    }
    #pragma unroll
    for (int nt = 0; nt < 2; ++nt) {
      int c = warpN + nt * 16 + lane16;
      #pragma unroll
      for (int j = 0; j < 8; ++j) bu[nt].u[j] = sB[(j < 4 ? hi * 4 + j : 4 + hi * 4 + j)][c];
    }
    #pragma unroll
    for (int mt = 0; mt < 2; ++mt) {
      v16bf av = __builtin_bit_cast(v16bf, au[mt]);
      #pragma unroll
      for (int nt = 0; nt < 2; ++nt) {
        v16bf bv = __builtin_bit_cast(v16bf, bu[nt]);
        acc[mt][nt] = __builtin_amdgcn_wmma_f32_16x16x32_bf16(
            false, av, false, bv, (short)0, acc[mt][nt], false, false);
      }
    }
    __syncthreads();
  }

  #pragma unroll
  for (int mt = 0; mt < 2; ++mt)
    #pragma unroll
    for (int nt = 0; nt < 2; ++nt)
      #pragma unroll
      for (int r = 0; r < 8; ++r) {
        int gm = mbase + warpM + mt * 16 + hi * 8 + r;
        int gn = nbase + warpN + nt * 16 + lane16;
        float v = acc[mt][nt][r];
        if constexpr (MODE == 3) {
          ((float*)Out)[ooff + (size_t)gm * ldo + gn] = v;
        } else {
          ((unsigned short*)Out)[ooff + (size_t)gm * ldo + gn] = f2bf(v);
        }
      }
}

// ---------------------------------------------------------------------------
// LayerNorm over D=768, one block (256 threads) per row; bf16 output.
// ---------------------------------------------------------------------------
__global__ void __launch_bounds__(256) ln_kernel(
    const float* __restrict__ X, const float* __restrict__ w,
    const float* __restrict__ b, unsigned short* __restrict__ Out)
{
  __shared__ float red[256];
  int row = blockIdx.x;
  int t = threadIdx.x;
  const float* xr = X + (size_t)row * 768;
  float v0 = xr[t], v1 = xr[t + 256], v2 = xr[t + 512];
  red[t] = v0 + v1 + v2; __syncthreads();
  for (int o = 128; o > 0; o >>= 1) { if (t < o) red[t] += red[t + o]; __syncthreads(); }
  float mean = red[0] * (1.f / 768.f);
  __syncthreads();
  float d0 = v0 - mean, d1 = v1 - mean, d2 = v2 - mean;
  red[t] = d0 * d0 + d1 * d1 + d2 * d2; __syncthreads();
  for (int o = 128; o > 0; o >>= 1) { if (t < o) red[t] += red[t + o]; __syncthreads(); }
  float rstd = rsqrtf(red[0] * (1.f / 768.f) + 1e-6f);
  unsigned short* orow = Out + (size_t)row * 768;
  orow[t]       = f2bf(d0 * rstd * w[t]       + b[t]);
  orow[t + 256] = f2bf(d1 * rstd * w[t + 256] + b[t + 256]);
  orow[t + 512] = f2bf(d2 * rstd * w[t + 512] + b[t + 512]);
}

// ---------------------------------------------------------------------------
// Softmax over key dim (256) with rel-pos bias gather; probs out as bf16.
// ---------------------------------------------------------------------------
__global__ void __launch_bounds__(256) softmax_kernel(
    const float* __restrict__ S, unsigned short* __restrict__ P,
    const float* __restrict__ rpb, int l)
{
  __shared__ float red[256];
  int n  = blockIdx.x;
  int bh = blockIdx.y;
  int h  = bh % 12;
  int m  = threadIdx.x;
  int ih = (n >> 4) - (m >> 4) + 15;
  int iw = (n & 15) - (m & 15) + 15;
  float bias = rpb[(((size_t)l * 31 + ih) * 31 + iw) * 12 + h];
  size_t base = (size_t)bh * 65536 + (size_t)n * 256;
  float v = S[base + m] * 0.125f + bias;   // scale = HD^-0.5 = 1/8
  red[m] = v; __syncthreads();
  for (int o = 128; o > 0; o >>= 1) { if (m < o) red[m] = fmaxf(red[m], red[m + o]); __syncthreads(); }
  float mx = red[0]; __syncthreads();
  float e = __expf(v - mx);
  red[m] = e; __syncthreads();
  for (int o = 128; o > 0; o >>= 1) { if (m < o) red[m] += red[m + o]; __syncthreads(); }
  float inv = 1.0f / red[0];
  P[base + m] = f2bf(e * inv);
}

// ---------------------------------------------------------------------------
// Small fp32 GEMM (time-embedding paths, M=32): one output per thread.
// ---------------------------------------------------------------------------
template<int ACT>
__global__ void smallgemm(const float* __restrict__ A, const float* __restrict__ Bm,
                          const float* __restrict__ bias, float* __restrict__ Out,
                          int M, int N, int K)
{
  int idx = blockIdx.x * blockDim.x + threadIdx.x;
  if (idx >= M * N) return;
  int m = idx / N, n = idx % N;
  float acc = bias ? bias[n] : 0.f;
  for (int k = 0; k < K; ++k) acc += A[(size_t)m * K + k] * Bm[(size_t)k * N + n];
  if (ACT) acc = acc / (1.f + __expf(-acc));
  Out[idx] = acc;
}

// Sinusoidal time embedding -> emb[32][256]
__global__ void time_emb_kernel(const float* __restrict__ t, float* __restrict__ emb)
{
  int idx = blockIdx.x * blockDim.x + threadIdx.x;
  if (idx >= 32 * 256) return;
  int b = idx >> 8; int j = idx & 255;
  int i = (j < 128) ? j : j - 128;
  float freq = __expf(-logf(10000.f) * (float)i / 127.f);
  float a = t[b] * freq;
  emb[idx] = (j < 128) ? sinf(a) : cosf(a);
}

// Patch embed: x_s[b,n,d] = sum_{c,p,q} x[b,c,2i+p,2j+q]*pw[d,c,p,q] + pb[d] + pos[n,d]
__global__ void patch_embed_kernel(const float* __restrict__ x, const float* __restrict__ pw,
                                   const float* __restrict__ pb, const float* __restrict__ pos,
                                   float* __restrict__ xs)
{
  int idx = blockIdx.x * blockDim.x + threadIdx.x;
  if (idx >= 32 * 256 * 768) return;
  int d = idx % 768;
  int n = (idx / 768) & 255;
  int b = idx / (768 * 256);
  int i = n >> 4, j = n & 15;
  float acc = pb[d] + pos[(size_t)n * 768 + d];
  const float* wd = pw + (size_t)d * 16;
  #pragma unroll
  for (int c = 0; c < 4; ++c)
    #pragma unroll
    for (int p = 0; p < 2; ++p)
      #pragma unroll
      for (int q = 0; q < 2; ++q)
        acc += x[(((size_t)b * 4 + c) * 32 + (i * 2 + p)) * 32 + (j * 2 + q)] * wd[c * 4 + p * 2 + q];
  xs[idx] = acc;
}

// Decoder (D->16) + unpatchify to [B,C,32,32]
__global__ void decoder_kernel(const unsigned short* __restrict__ hln,
                               const float* __restrict__ dw, const float* __restrict__ db,
                               float* __restrict__ out)
{
  int idx = blockIdx.x * blockDim.x + threadIdx.x;
  if (idx >= 32 * 256 * 16) return;
  int e = idx & 15;
  int n = (idx >> 4) & 255;
  int b = idx >> 12;
  float acc = db[e];
  const unsigned short* hr = hln + ((size_t)b * 256 + n) * 768;
  for (int d = 0; d < 768; ++d) acc += bf2f(hr[d]) * dw[(size_t)d * 16 + e];
  int c = e >> 2, p = (e >> 1) & 1, q = e & 1;
  int i = n >> 4, j = n & 15;
  out[(((size_t)b * 4 + c) * 32 + (i * 2 + p)) * 32 + (j * 2 + q)] = acc;
}

// ---------------------------------------------------------------------------
extern "C" void kernel_launch(void* const* d_in, const int* in_sizes, int n_in,
                              void* d_out, int out_size, void* d_ws, size_t ws_size,
                              hipStream_t stream)
{
  (void)in_sizes; (void)n_in; (void)out_size; (void)ws_size;

  const float* x       = (const float*)d_in[0];
  const float* t       = (const float*)d_in[1];
  const float* patch_w = (const float*)d_in[2];
  const float* patch_b = (const float*)d_in[3];
  const float* pos     = (const float*)d_in[4];
  const float* t_w1    = (const float*)d_in[5];
  const float* t_b1    = (const float*)d_in[6];
  const float* t_w2    = (const float*)d_in[7];
  const float* t_b2    = (const float*)d_in[8];
  const float* norm1_w = (const float*)d_in[9];
  const float* norm1_b = (const float*)d_in[10];
  const float* wqs     = (const float*)d_in[11];
  const float* bqs     = (const float*)d_in[12];
  const float* wqt     = (const float*)d_in[13];
  const float* bqt     = (const float*)d_in[14];
  const float* wks     = (const float*)d_in[15];
  const float* bks     = (const float*)d_in[16];
  const float* wkt     = (const float*)d_in[17];
  const float* bkt     = (const float*)d_in[18];
  const float* wvs     = (const float*)d_in[19];
  const float* bvs     = (const float*)d_in[20];
  const float* wvt     = (const float*)d_in[21];
  const float* bvt     = (const float*)d_in[22];
  const float* proj_w  = (const float*)d_in[23];
  const float* proj_b  = (const float*)d_in[24];
  const float* rpb     = (const float*)d_in[25];
  const float* norm2_w = (const float*)d_in[26];
  const float* norm2_b = (const float*)d_in[27];
  const float* mlp_w1  = (const float*)d_in[28];
  const float* mlp_b1  = (const float*)d_in[29];
  const float* mlp_w2  = (const float*)d_in[30];
  const float* mlp_b2  = (const float*)d_in[31];
  const float* norm_w  = (const float*)d_in[32];
  const float* norm_b  = (const float*)d_in[33];
  const float* dec_w   = (const float*)d_in[34];
  const float* dec_b   = (const float*)d_in[35];

  char* ws = (char*)d_ws;
  size_t off = 0;
  auto alloc = [&](size_t bytes) -> void* {
    void* p = ws + off; off += (bytes + 255) & ~(size_t)255; return p;
  };
  float*          x_s   = (float*)alloc(8192ull * 768 * 4);
  unsigned short* hln   = (unsigned short*)alloc(8192ull * 768 * 2);
  unsigned short* qb    = (unsigned short*)alloc(8192ull * 768 * 2);
  unsigned short* kbuf  = (unsigned short*)alloc(8192ull * 768 * 2);
  unsigned short* vbuf  = (unsigned short*)alloc(8192ull * 768 * 2);
  unsigned short* obuf  = (unsigned short*)alloc(8192ull * 768 * 2);
  float*          score = (float*)alloc(384ull * 65536 * 4);
  unsigned short* probs = (unsigned short*)alloc(384ull * 65536 * 2);
  unsigned short* hid   = (unsigned short*)alloc(8192ull * 3072 * 2);
  float*          emb   = (float*)alloc(32ull * 256 * 4);
  float*          tt    = (float*)alloc(32ull * 768 * 4);
  float*          xt    = (float*)alloc(32ull * 768 * 4);
  float*          tq    = (float*)alloc(32ull * 768 * 4);
  float*          tk    = (float*)alloc(32ull * 768 * 4);
  float*          tv    = (float*)alloc(32ull * 768 * 4);

  patch_embed_kernel<<<(32 * 256 * 768) / 256, 256, 0, stream>>>(x, patch_w, patch_b, pos, x_s);
  time_emb_kernel<<<32, 256, 0, stream>>>(t, emb);
  smallgemm<1><<<96, 256, 0, stream>>>(emb, t_w1, t_b1, tt, 32, 768, 256);
  smallgemm<0><<<96, 256, 0, stream>>>(tt, t_w2, t_b2, xt, 32, 768, 768);

  dim3 g0(64, 12, 1);    // [8192 x 768] out
  dim3 g1(64, 48, 1);    // [8192 x 3072] out
  dim3 gs(2, 4, 384);    // scores per (b,h): 256x256
  dim3 gpv(2, 1, 384);   // P*V per (b,h): 256x64
  dim3 gsm(256, 384, 1);

  for (int l = 0; l < 12; ++l) {
    ln_kernel<<<8192, 256, 0, stream>>>(x_s, norm1_w + l * 768, norm1_b + l * 768, hln);

    smallgemm<0><<<96, 256, 0, stream>>>(xt, wqt + (size_t)l * 768 * 768, bqt + l * 768, tq, 32, 768, 768);
    smallgemm<0><<<96, 256, 0, stream>>>(xt, wkt + (size_t)l * 768 * 768, bkt + l * 768, tk, 32, 768, 768);
    smallgemm<0><<<96, 256, 0, stream>>>(xt, wvt + (size_t)l * 768 * 768, bvt + l * 768, tv, 32, 768, 768);

    gemm_wmma_w<0><<<g0, 256, 0, stream>>>(hln, wqs + (size_t)l * 768 * 768, bqs + l * 768, tq, qb,   768, 768, 768, 768);
    gemm_wmma_w<0><<<g0, 256, 0, stream>>>(hln, wks + (size_t)l * 768 * 768, bks + l * 768, tk, kbuf, 768, 768, 768, 768);
    gemm_wmma_w<0><<<g0, 256, 0, stream>>>(hln, wvs + (size_t)l * 768 * 768, bvs + l * 768, tv, vbuf, 768, 768, 768, 768);

    gemm_wmma_att<3><<<gs, 256, 0, stream>>>(qb, kbuf, score, 64, 768, 768, 256);
    softmax_kernel<<<gsm, 256, 0, stream>>>(score, probs, rpb, l);
    gemm_wmma_att<4><<<gpv, 256, 0, stream>>>(probs, vbuf, obuf, 256, 256, 768, 768);

    gemm_wmma_w<1><<<g0, 256, 0, stream>>>(obuf, proj_w + (size_t)l * 768 * 768, proj_b + l * 768, nullptr, x_s, 768, 768, 768, 768);

    ln_kernel<<<8192, 256, 0, stream>>>(x_s, norm2_w + l * 768, norm2_b + l * 768, hln);
    gemm_wmma_w<2><<<g1, 256, 0, stream>>>(hln, mlp_w1 + (size_t)l * 768 * 3072, mlp_b1 + l * 3072, nullptr, hid, 768, 768, 3072, 3072);
    gemm_wmma_w<1><<<g0, 256, 0, stream>>>(hid, mlp_w2 + (size_t)l * 3072 * 768, mlp_b2 + l * 768, nullptr, x_s, 3072, 3072, 768, 768);
  }

  ln_kernel<<<8192, 256, 0, stream>>>(x_s, norm_w, norm_b, hln);
  decoder_kernel<<<(32 * 256 * 16) / 256, 256, 0, stream>>>(hln, dec_w, dec_b, (float*)d_out);
}
